// GRUBaseline_44049184587814
// MI455X (gfx1250) — matrix operand order repaired and checked by reading, not compiled, not run on hardware
//
#include <hip/hip_runtime.h>
#include <hip/hip_bf16.h>

#define B_    1024
#define T_    1024
#define OBS_  16
#define HID_  128
#define TSTEPS (T_ - 1)     // 1023
#define NTHREADS 384        // 12 waves (wave32)

typedef __attribute__((ext_vector_type(16))) __bf16 v16bf;
typedef __attribute__((ext_vector_type(8)))  float  v8f;

// ---- WMMA fragment helpers (CDNA5 16x16x32 bf16, contiguous k order: elem e of lane half h -> K=16h+e) ----

// A-matrix 16x32 bf16 from an LDS row-major [16][ldk] tile, K-chunk base kb.
// lanes 0-15: row M=lane,    elems 0..15 -> K=kb+0..15
// lanes 16-31: row M=lane-16, elems 0..15 -> K=kb+16..31
__device__ __forceinline__ v16bf load_a_frag(const __bf16* base, int ldk, int kb, int lane) {
  int row = lane & 15;
  int hi  = (lane >> 4) & 1;
  const __bf16* p = base + row * ldk + kb + hi * 16;
  v16bf a;
#pragma unroll
  for (int e = 0; e < 8; ++e) a[e] = p[e];
#pragma unroll
  for (int e = 0; e < 8; ++e) a[8 + e] = p[8 + e];
  return a;
}

// B-matrix 32x16 bf16 from a global f32 weight W[n][k] (row-major, ld = HID_):
// B[k][n] = W[col0+n][kb+k].  lanes 0-15: K=kb+0..15 (elem e -> K=kb+e), N=lane&15;
// lanes 16-31: K=kb+16..31.
__device__ __forceinline__ v16bf load_bh_frag(const float* W, int col0, int kb, int lane) {
  int n = lane & 15;
  int kbase = kb + (((lane >> 4) & 1) << 4);
  const float* p = W + (size_t)(col0 + n) * HID_;
  v16bf b;
#pragma unroll
  for (int e = 0; e < 16; ++e) b[e] = (__bf16)p[kbase + e];
  return b;
}

// x-projection B fragment: K=17 real weights, K=17 slot carries the bias
// (paired with a constant 1.0 column in the A tile), K=18..31 zero.
__device__ __forceinline__ v16bf load_bx_frag(const float* W_ih, float bias,
                                              int col0, int lane) {
  int n = lane & 15;
  int kbase = (((lane >> 4) & 1) << 4);
  const float* p = W_ih + (size_t)(col0 + n) * (OBS_ + 1);
  v16bf b;
#pragma unroll
  for (int e = 0; e < 16; ++e) {
    int k = kbase + e;
    float v = (k < OBS_ + 1) ? p[k] : ((k == OBS_ + 1) ? bias : 0.0f);
    b[e] = (__bf16)v;
  }
  return b;
}

__device__ __forceinline__ float sigmoidf_(float x) {
  return 1.0f / (1.0f + __expf(-x));
}

__device__ __forceinline__ v8f wmma_bf16(v16bf a, v16bf b, v8f c) {
  return __builtin_amdgcn_wmma_f32_16x16x32_bf16(false, a, false, b, (short)0, c,
                                                 false, false);
}

// One workgroup = 16 batch rows, 12 waves, runs the full 1023-step recurrence.
// Waves 0-7: r/z gate columns 0..255 (merged x+h accumulator).
// Waves 8-11: n gate columns 256..383 (separate xn / hn accumulators).
// Input staging is software-pipelined: step t+1's data is loaded into
// registers during step t and committed to LDS at the top of step t+1.
__global__ __launch_bounds__(NTHREADS) void gru_seq_kernel(
    const float* __restrict__ obs,   const float* __restrict__ times,
    const float* __restrict__ W_ih,  const float* __restrict__ b_ih,
    const float* __restrict__ W_hh,  const float* __restrict__ b_hh,
    const float* __restrict__ W_out, const float* __restrict__ b_out,
    float* __restrict__ pred)
{
  __shared__ float  sh_h [16][HID_];      // fp32 master hidden state (8 KB)
  __shared__ __bf16 sh_hb[16][HID_];      // bf16 copy for WMMA A operand (4 KB)
  __shared__ __bf16 sh_x [16][32];        // input tile: obs, dt, 1.0, pad (1 KB)
  __shared__ float  sh_rz[16][2 * HID_];  // pre-act r|z (x+h+biases) (16 KB)
  __shared__ float  sh_xn[16][HID_];      // xn + b_ih_n (8 KB)
  __shared__ float  sh_hn[16][HID_];      // hn (no bias) (8 KB)
  __shared__ float  sh_bhh_n[HID_];       // b_hh n-section (fp32 exact)

  const int tid  = threadIdx.x;
  const int lane = tid & 31;
  const int wave = tid >> 5;              // 0..11
  const int b0   = blockIdx.x * 16;
  const int nn   = lane & 15;
  const bool is_n_wave = (wave >= 8);

  // ---- init: h = 0, input-pad columns (17 -> 1.0 bias column, 18..31 -> 0) ----
  for (int i = tid; i < 16 * HID_; i += NTHREADS) {
    sh_h[0][i]  = 0.0f;
    sh_hb[0][i] = (__bf16)0.0f;
  }
  for (int i = tid; i < 16 * 15; i += NTHREADS) {
    int m = i / 15, c = 17 + (i % 15);
    sh_x[m][c] = (c == 17) ? (__bf16)1.0f : (__bf16)0.0f;
  }
  if (tid < HID_) sh_bhh_n[tid] = b_hh[2 * HID_ + tid];

  // ---- preload weight B-fragments into VGPRs (reused for all 1023 steps) ----
  v16bf wx_f[2];
  v16bf wh_f[2][4];
#pragma unroll
  for (int i = 0; i < 2; ++i) {
    const int col0 = (wave * 2 + i) * 16;
    const float bias = b_ih[col0 + nn] + (is_n_wave ? 0.0f : b_hh[col0 + nn]);
    wx_f[i] = load_bx_frag(W_ih, bias, col0, lane);
#pragma unroll
    for (int kc = 0; kc < 4; ++kc)
      wh_f[i][kc] = load_bh_frag(W_hh, col0, kc * 32, lane);
  }
  v16bf wout_f[4] = {};
  float bout_v = 0.0f;
  if (wave == 0) {
#pragma unroll
    for (int kc = 0; kc < 4; ++kc)
      wout_f[kc] = load_bh_frag(W_out, 0, kc * 32, lane);
    bout_v = b_out[nn];
  }

  // ---- staging pipeline registers: preload step 0 ----
  const bool is_obs_thr = (tid < 256);
  const bool is_dt_thr  = (tid >= 256 && tid < 272);
  const int  sm = is_obs_thr ? (tid >> 4) : (tid - 256);  // staged batch row
  const int  so = tid & 15;                               // staged obs column
  const float* obs_src  = obs + ((size_t)(b0 + sm) * T_) * OBS_ + so;
  const float* time_src = times + (size_t)(b0 + sm) * T_;

  float rx = 0.0f, tprev = 0.0f, tnext = 0.0f;
  if (is_obs_thr) {
    rx = obs_src[0];
  } else if (is_dt_thr) {
    tprev = time_src[0];
    tnext = time_src[1];
  }

  for (int t = 0; t < TSTEPS; ++t) {
    // ---- commit staged step-t input to LDS ----
    if (is_obs_thr)     sh_x[sm][so] = (__bf16)rx;
    else if (is_dt_thr) sh_x[sm][16] = (__bf16)(tnext - tprev);
    __syncthreads();

    // ---- issue step t+1 loads now; they have the whole step to complete ----
    if (t + 1 < TSTEPS) {
      if (is_obs_thr) {
        rx = obs_src[(size_t)(t + 1) * OBS_];
      } else if (is_dt_thr) {
        tprev = tnext;
        tnext = time_src[t + 2];
      }
    }

    // ---- shared A-fragments for both gate branches ----
    const v16bf ax  = load_a_frag(&sh_x[0][0], 32, 0, lane);
    const v16bf ah0 = load_a_frag(&sh_hb[0][0], HID_, 0,  lane);
    const v16bf ah1 = load_a_frag(&sh_hb[0][0], HID_, 32, lane);
    const v16bf ah2 = load_a_frag(&sh_hb[0][0], HID_, 64, lane);
    const v16bf ah3 = load_a_frag(&sh_hb[0][0], HID_, 96, lane);
    const int mb = (lane >> 4) ? 8 : 0;

    if (!is_n_wave) {
      // r/z: single accumulator = x@W_ih^T + h@W_hh^T + (b_ih+b_hh)
      v8f a0 = {}, a1 = {};
      a0 = wmma_bf16(ax,  wx_f[0],    a0);
      a1 = wmma_bf16(ax,  wx_f[1],    a1);
      a0 = wmma_bf16(ah0, wh_f[0][0], a0);
      a1 = wmma_bf16(ah0, wh_f[1][0], a1);
      a0 = wmma_bf16(ah1, wh_f[0][1], a0);
      a1 = wmma_bf16(ah1, wh_f[1][1], a1);
      a0 = wmma_bf16(ah2, wh_f[0][2], a0);
      a1 = wmma_bf16(ah2, wh_f[1][2], a1);
      a0 = wmma_bf16(ah3, wh_f[0][3], a0);
      a1 = wmma_bf16(ah3, wh_f[1][3], a1);
      const int c0 = (wave * 2) * 16 + nn;
#pragma unroll
      for (int v = 0; v < 8; ++v) {
        sh_rz[mb + v][c0]      = a0[v];
        sh_rz[mb + v][c0 + 16] = a1[v];
      }
    } else {
      // n: keep xn and hn separate (n = tanh(xn + r*hn))
      v8f x0 = {}, x1 = {}, h0 = {}, h1 = {};
      x0 = wmma_bf16(ax,  wx_f[0],    x0);
      x1 = wmma_bf16(ax,  wx_f[1],    x1);
      h0 = wmma_bf16(ah0, wh_f[0][0], h0);
      h1 = wmma_bf16(ah0, wh_f[1][0], h1);
      h0 = wmma_bf16(ah1, wh_f[0][1], h0);
      h1 = wmma_bf16(ah1, wh_f[1][1], h1);
      h0 = wmma_bf16(ah2, wh_f[0][2], h0);
      h1 = wmma_bf16(ah2, wh_f[1][2], h1);
      h0 = wmma_bf16(ah3, wh_f[0][3], h0);
      h1 = wmma_bf16(ah3, wh_f[1][3], h1);
      const int c0 = (wave * 2 - 16) * 16 + nn;   // local n-column 0..127
#pragma unroll
      for (int v = 0; v < 8; ++v) {
        sh_xn[mb + v][c0]      = x0[v];
        sh_xn[mb + v][c0 + 16] = x1[v];
        sh_hn[mb + v][c0]      = h0[v];
        sh_hn[mb + v][c0 + 16] = h1[v];
      }
    }
    __syncthreads();

    // ---- elementwise GRU gate update: 16*128 elems over 384 threads ----
    for (int idx = tid; idx < 16 * HID_; idx += NTHREADS) {
      const int m = idx >> 7, j = idx & (HID_ - 1);
      const float r = sigmoidf_(sh_rz[m][j]);
      const float z = sigmoidf_(sh_rz[m][HID_ + j]);
      const float n = tanhf(sh_xn[m][j] + r * (sh_hn[m][j] + sh_bhh_n[j]));
      const float hnew = n + z * (sh_h[m][j] - n);   // == (1-z)*n + z*h
      sh_h[m][j]  = hnew;
      sh_hb[m][j] = (__bf16)hnew;
    }
    __syncthreads();

    // ---- output projection pred[:,t] = h @ W_out^T + b_out (wave 0, K=128) ----
    if (wave == 0) {
      v8f cp = {};
#pragma unroll
      for (int kc = 0; kc < 4; ++kc) {
        v16bf ap = load_a_frag(&sh_hb[0][0], HID_, kc * 32, lane);
        cp = wmma_bf16(ap, wout_f[kc], cp);
      }
#pragma unroll
      for (int v = 0; v < 8; ++v)
        pred[((size_t)(b0 + mb + v) * TSTEPS + t) * OBS_ + nn] = cp[v] + bout_v;
    }
    // next iteration's post-staging barrier closes the step: the pred phase
    // only reads sh_hb, which is next written after that barrier.
  }
}

// target = observed_values[:, 1:, :]  — pure bandwidth copy, float4 vectorized.
__global__ void copy_target_kernel(const float4* __restrict__ obs4,
                                   float4* __restrict__ out4)
{
  const long total = (long)B_ * TSTEPS * (OBS_ / 4);
  for (long i = (long)blockIdx.x * blockDim.x + threadIdx.x; i < total;
       i += (long)gridDim.x * blockDim.x) {
    const long row = i >> 2;            // b*1023 + t
    const int  o4  = (int)(i & 3);
    const long b   = row / TSTEPS;
    const long t   = row - b * TSTEPS;
    out4[i] = obs4[((b * T_) + t + 1) * 4 + o4];
  }
}

extern "C" void kernel_launch(void* const* d_in, const int* in_sizes, int n_in,
                              void* d_out, int out_size, void* d_ws, size_t ws_size,
                              hipStream_t stream) {
  const float* obs   = (const float*)d_in[0];
  const float* times = (const float*)d_in[1];
  const float* W_ih  = (const float*)d_in[2];
  const float* b_ih  = (const float*)d_in[3];
  const float* W_hh  = (const float*)d_in[4];
  const float* b_hh  = (const float*)d_in[5];
  const float* W_out = (const float*)d_in[6];
  const float* b_out = (const float*)d_in[7];

  float* pred   = (float*)d_out;
  float* target = pred + (size_t)B_ * TSTEPS * OBS_;

  copy_target_kernel<<<4096, 256, 0, stream>>>((const float4*)obs, (float4*)target);
  gru_seq_kernel<<<B_ / 16, NTHREADS, 0, stream>>>(obs, times, W_ih, b_ih, W_hh,
                                                   b_hh, W_out, b_out, pred);
}
